// Transformer_with_norm_49572512530940
// MI455X (gfx1250) — matrix operand order repaired and checked by reading, not compiled, not run on hardware
//
#include <hip/hip_runtime.h>
#include <hip/hip_bf16.h>

// ---------------------------------------------------------------------------
// Problem constants (from reference): B=4, N=4096, IN_CH=128, DIM=256, K=8,
// PHD=64, HID=1024.
// ---------------------------------------------------------------------------
#define NPTS 4096
#define BATCH 4
#define INC 128
#define DIMC 256
#define KNN 8
#define PHDC 64
#define HIDC 1024
#define BN   16384      // BATCH*NPTS
#define BNK  131072     // BATCH*NPTS*KNN

typedef __attribute__((ext_vector_type(16))) __bf16 v16bf;
typedef __attribute__((ext_vector_type(8)))  float  v8f;

union Frag16 { v16bf v; unsigned short u[16]; unsigned w[8]; };

__device__ __forceinline__ unsigned short f2bf(float f) {
  union { float f; unsigned u; } x; x.f = f;
  unsigned r = x.u + 0x7FFFu + ((x.u >> 16) & 1u);   // round-to-nearest-even
  return (unsigned short)(r >> 16);
}
__device__ __forceinline__ float bf2f(unsigned short h) {
  union { unsigned u; float f; } x; x.u = ((unsigned)h) << 16;
  return x.f;
}
__device__ __forceinline__ unsigned pack2bf(float lo, float hi) {
  return (unsigned)f2bf(lo) | ((unsigned)f2bf(hi) << 16);
}
// A-matrix (16x32 bf16) per-lane K base for dword pair p (ISA 7.12.2):
// lanes 0-15 hold K={0..7,16..23}; lanes 16-31 hold K={8..15,24..31}.
__device__ __forceinline__ int kbaseA(int p, int g) {
  return ((p < 4) ? 0 : 16) + g * 8 + (p & 3) * 2;
}

// ---------------------------------------------------------------------------
// Generic WMMA GEMM: Out[M,Nn] = A[M,Kd] * X[Kd,Nn] + bias (+ residual).
// A and X are bf16 (pre-converted); accumulate f32. A fragment = 8 dword
// loads (K-pairs adjacent). One 16x16 tile per wave; 4 waves per block.
// grid = (Nn/16, M/64, batch)
// ---------------------------------------------------------------------------
template<bool OBF16, bool HAS_RES>
__global__ __launch_bounds__(128) void wmma_gemm(
    const unsigned short* __restrict__ A, const unsigned short* __restrict__ X,
    void* __restrict__ Ov, const float* __restrict__ bias,
    const float* __restrict__ resid,
    int M, int Kd, long long Nn, long long strideX, long long strideO)
{
  const int lane = threadIdx.x & 31;
  const int w    = threadIdx.x >> 5;
  const int g    = lane >> 4;
  const int mn   = lane & 15;
  const int tileM = blockIdx.y * 4 + w;
  const long long col = (long long)blockIdx.x * 16 + mn;
  const int row = tileM * 16 + mn;
  const long long bx = (long long)blockIdx.z * strideX;
  const long long bo = (long long)blockIdx.z * strideO;
  const unsigned short* Xb = X + bx;

  v8f acc = {};
  for (int kk = 0; kk < Kd; kk += 32) {
    Frag16 fa, fb;
#pragma unroll
    for (int p = 0; p < 8; ++p)      // dword loads: two adjacent K per dword
      fa.w[p] = *(const unsigned*)(A + (long long)row * Kd + kk + g * 16 + 2 * p);
#pragma unroll
    for (int e = 0; e < 16; ++e)     // B: K = g*16 + e, column-strided
      fb.u[e] = Xb[(long long)(kk + g * 16 + e) * Nn + col];
    if (kk + 32 < Kd)
      __builtin_prefetch(&A[(long long)row * Kd + kk + 32], 0, 1);
    acc = __builtin_amdgcn_wmma_f32_16x16x32_bf16(false, fa.v, false, fb.v,
                                                  (short)0, acc, false, false);
  }
#pragma unroll
  for (int r = 0; r < 8; ++r) {
    int m = tileM * 16 + g * 8 + r;                 // D: VGPR r -> M = r + 8*g
    float val = acc[r];
    if (bias) val += bias[m];
    long long oi = (long long)m * Nn + (long long)blockIdx.x * 16 + mn;
    if (HAS_RES) val += resid[bo + oi];
    if (OBF16) ((unsigned short*)Ov)[bo + oi] = f2bf(val);
    else       ((float*)Ov)[bo + oi] = val;
  }
}

// ---------------------------------------------------------------------------
// f32 -> bf16 bulk conversion.
// ---------------------------------------------------------------------------
__global__ __launch_bounds__(256) void cvt_bf16_kernel(
    const float* __restrict__ src, unsigned short* __restrict__ dst, long long n)
{
  long long i = (long long)blockIdx.x * 256 + threadIdx.x;
  if (i < n) dst[i] = f2bf(src[i]);
}

// ---------------------------------------------------------------------------
// Brute-force KNN (K=8 incl. self). One thread per (b,n), insertion top-8.
// Neighbor order does not affect the output (all k-reductions are symmetric).
// ---------------------------------------------------------------------------
__global__ __launch_bounds__(256) void knn_kernel(const float* __restrict__ pos,
                                                  int* __restrict__ idx)
{
  int t = blockIdx.x * 256 + threadIdx.x;
  int b = t >> 12, n = t & 4095;
  const float* pb = pos + (long long)b * 3 * NPTS;
  float px = pb[n], py = pb[NPTS + n], pz = pb[2 * NPTS + n];
  float dist[KNN]; int id[KNN];
#pragma unroll
  for (int i = 0; i < KNN; ++i) { dist[i] = 3.4e38f; id[i] = n; }
  for (int m = 0; m < NPTS; ++m) {
    float dx = px - pb[m], dy = py - pb[NPTS + m], dz = pz - pb[2 * NPTS + m];
    float d = dx * dx + dy * dy + dz * dz;
    if (d < dist[KNN - 1]) {
      dist[KNN - 1] = d; id[KNN - 1] = m;
#pragma unroll
      for (int j = KNN - 1; j > 0; --j) {
        if (dist[j] < dist[j - 1]) {
          float td = dist[j]; dist[j] = dist[j - 1]; dist[j - 1] = td;
          int ti = id[j];    id[j]   = id[j - 1];   id[j - 1]   = ti;
        }
      }
    }
  }
#pragma unroll
  for (int i = 0; i < KNN; ++i) idx[(long long)t * KNN + i] = id[i];
}

// ---------------------------------------------------------------------------
// Fold eval-BN into per-channel scale/shift; build Wcat = [W2_pm | W2_nm] bf16.
// ---------------------------------------------------------------------------
__global__ void prep_kernel(
    const float* pm_b1, const float* pm_g, const float* pm_be, const float* pm_m,
    const float* pm_v, const float* pm_W2, const float* pm_b2,
    const float* nm_b1, const float* nm_g, const float* nm_be, const float* nm_m,
    const float* nm_v, const float* nm_W2, const float* nm_b2,
    const float* am_b1, const float* am_g, const float* am_be, const float* am_m,
    const float* am_v,
    float* s_pm, float* sh_pm, float* s_nm, float* sh_nm,
    float* s_am, float* sh_am, unsigned short* wcat, float* bcat)
{
  int t = threadIdx.x;
  if (t < PHDC) {
    float s = pm_g[t] * rsqrtf(pm_v[t] + 1e-5f);
    s_pm[t] = s; sh_pm[t] = s * pm_b1[t] + pm_be[t] - pm_m[t] * s;
    float s2 = nm_g[t] * rsqrtf(nm_v[t] + 1e-5f);
    s_nm[t] = s2; sh_nm[t] = s2 * nm_b1[t] + nm_be[t] - nm_m[t] * s2;
  }
  if (t < HIDC) {
    float s = am_g[t] * rsqrtf(am_v[t] + 1e-5f);
    s_am[t] = s; sh_am[t] = s * am_b1[t] + am_be[t] - am_m[t] * s;
  }
  if (t < DIMC) bcat[t] = pm_b2[t] + nm_b2[t];
  for (int i = t; i < DIMC * 2 * PHDC; i += 1024) {
    int m = i >> 7, k = i & 127;
    wcat[i] = f2bf((k < PHDC) ? pm_W2[m * PHDC + k] : nm_W2[m * PHDC + (k - PHDC)]);
  }
}

// ---------------------------------------------------------------------------
// pos/norm relative 3->64 MLP1 (+BN+relu), write bf16 hcat (128 x BNK).
// ---------------------------------------------------------------------------
__global__ __launch_bounds__(256) void rel_mlp1_kernel(
    const float* __restrict__ pos, const float* __restrict__ nrm,
    const int* __restrict__ idx,
    const float* __restrict__ pmW1, const float* __restrict__ nmW1,
    const float* __restrict__ s_pm, const float* __restrict__ sh_pm,
    const float* __restrict__ s_nm, const float* __restrict__ sh_nm,
    unsigned short* __restrict__ hcat)
{
  int t = blockIdx.x * 256 + threadIdx.x;        // 0..BNK-1
  int b = t >> 15;                                // 32768 = NPTS*KNN
  int nk = t & 32767;
  int n = nk >> 3;
  int j = idx[t];
  const float* pb = pos + (long long)b * 3 * NPTS;
  const float* nb = nrm + (long long)b * 3 * NPTS;
  float r0 = pb[n] - pb[j], r1 = pb[NPTS + n] - pb[NPTS + j],
        r2 = pb[2 * NPTS + n] - pb[2 * NPTS + j];
  float q0 = nb[n] - nb[j], q1 = nb[NPTS + n] - nb[NPTS + j],
        q2 = nb[2 * NPTS + n] - nb[2 * NPTS + j];
  for (int c = 0; c < PHDC; ++c) {
    float ap = pmW1[c * 3] * r0 + pmW1[c * 3 + 1] * r1 + pmW1[c * 3 + 2] * r2;
    hcat[(long long)c * BNK + t] = f2bf(fmaxf(s_pm[c] * ap + sh_pm[c], 0.f));
    float an = nmW1[c * 3] * q0 + nmW1[c * 3 + 1] * q1 + nmW1[c * 3 + 2] * q2;
    hcat[(long long)(PHDC + c) * BNK + t] = f2bf(fmaxf(s_nm[c] * an + sh_nm[c], 0.f));
  }
}

// ---------------------------------------------------------------------------
// u[c][bnk] = query[b,c,n] - key[b,c,idx] + xyz[c][bnk]   (all bf16 storage)
// ---------------------------------------------------------------------------
__global__ __launch_bounds__(256) void build_u_kernel(
    const unsigned short* __restrict__ query, const unsigned short* __restrict__ keyb,
    const unsigned short* __restrict__ xyz, const int* __restrict__ idx,
    unsigned short* __restrict__ u)
{
  long long t = (long long)blockIdx.x * 256 + threadIdx.x;  // c*BNK + bnk
  int c = (int)(t >> 17);
  int bnk = (int)(t & (BNK - 1));
  int b = bnk >> 15, nk = bnk & 32767, n = nk >> 3;
  int j = idx[bnk];
  float q  = bf2f(query[((long long)b * DIMC + c) * NPTS + n]);
  float kv = bf2f(keyb [((long long)b * DIMC + c) * NPTS + j]);
  u[t] = f2bf(q - kv + bf2f(xyz[t]));
}

// ---------------------------------------------------------------------------
// Fused attention MLP: a = W2*(relu(bn(W1*u + b1))) + b2, per 16-point tile.
// LDS tiles stored [n][K] with padded stride so B fragments are dword
// ds-loads on distinct banks. 1024 WMMA per workgroup.
// ---------------------------------------------------------------------------
#define USTR 264    // 256 + 8 pad (ushorts)  -> lane n hits bank (4n+c)%64
#define HSTR 1032   // 1024 + 8 pad (ushorts)
__global__ __launch_bounds__(256) void am_fused_kernel(
    const unsigned short* __restrict__ W1, const unsigned short* __restrict__ W2,
    const float* __restrict__ s_am, const float* __restrict__ sh_am,
    const float* __restrict__ b2,
    const unsigned short* __restrict__ u, unsigned short* __restrict__ a)
{
  __shared__ unsigned short Us[16 * USTR];
  __shared__ unsigned short Hs[16 * HSTR];
  const int tid = threadIdx.x;
  const int w = tid >> 5;
  const int lane = tid & 31;
  const int g = lane >> 4, mn = lane & 15;
  const long long n0 = (long long)blockIdx.x * 16;

  for (int i = tid; i < DIMC * 16; i += 256) {
    int c = i >> 4, nn = i & 15;
    Us[nn * USTR + c] = u[(long long)c * BNK + n0 + nn];
  }
  __syncthreads();

  // layer 1: H(1024x16) = relu(scale*(W1 @ U) + shift), packed bf16 into LDS
  for (int rt = w * 8; rt < w * 8 + 8; ++rt) {
    v8f acc = {};
    for (int kk = 0; kk < DIMC; kk += 32) {
      Frag16 fa, fb;
#pragma unroll
      for (int p = 0; p < 8; ++p) {
        fa.w[p] = *(const unsigned*)(W1 + (long long)(rt * 16 + mn) * DIMC + kk + g * 16 + 2 * p);
        fb.w[p] = *(const unsigned*)(Us + mn * USTR + kk + g * 16 + 2 * p);
      }
      acc = __builtin_amdgcn_wmma_f32_16x16x32_bf16(false, fa.v, false, fb.v,
                                                    (short)0, acc, false, false);
    }
#pragma unroll
    for (int q = 0; q < 4; ++q) {
      int m = rt * 16 + g * 8 + 2 * q;
      float h0 = fmaxf(acc[2 * q]     * s_am[m]     + sh_am[m],     0.f);
      float h1 = fmaxf(acc[2 * q + 1] * s_am[m + 1] + sh_am[m + 1], 0.f);
      *(unsigned*)(Hs + mn * HSTR + m) = pack2bf(h0, h1);
    }
  }
  __syncthreads();

  // layer 2: a(256x16) = W2 @ H + b2
  for (int rt = w * 2; rt < w * 2 + 2; ++rt) {
    v8f acc = {};
    for (int kk = 0; kk < HIDC; kk += 32) {
      Frag16 fa, fb;
#pragma unroll
      for (int p = 0; p < 8; ++p) {
        fa.w[p] = *(const unsigned*)(W2 + (long long)(rt * 16 + mn) * HIDC + kk + g * 16 + 2 * p);
        fb.w[p] = *(const unsigned*)(Hs + mn * HSTR + kk + g * 16 + 2 * p);
      }
      acc = __builtin_amdgcn_wmma_f32_16x16x32_bf16(false, fa.v, false, fb.v,
                                                    (short)0, acc, false, false);
    }
#pragma unroll
    for (int r = 0; r < 8; ++r) {
      int m = rt * 16 + g * 8 + r;
      a[(long long)m * BNK + n0 + mn] = f2bf(acc[r] + b2[m]);
    }
  }
}

// ---------------------------------------------------------------------------
// softmax_one over k (e/(1+sum e), max-stabilized) and agg = sum attn*(v+xyz).
// One thread per (b,c,n); 8 bf16 loaded as one uint4. agg written as bf16.
// ---------------------------------------------------------------------------
__global__ __launch_bounds__(256) void softmax_agg_kernel(
    const unsigned short* __restrict__ a, const unsigned short* __restrict__ xyz,
    const unsigned short* __restrict__ value, unsigned short* __restrict__ agg)
{
  long long t = (long long)blockIdx.x * 256 + threadIdx.x;   // (b*256+c)*4096+n
  int n = (int)(t & 4095);
  int bc = (int)(t >> 12);
  int b = bc >> 8, c = bc & 255;
  long long base = ((long long)c << 17) + (((long long)b * NPTS + n) << 3);
  uint4 av4 = *(const uint4*)(a + base);
  uint4 xv4 = *(const uint4*)(xyz + base);
  unsigned au[4] = {av4.x, av4.y, av4.z, av4.w};
  unsigned xu[4] = {xv4.x, xv4.y, xv4.z, xv4.w};
  float av[8], xv[8];
#pragma unroll
  for (int i = 0; i < 4; ++i) {
    av[2 * i]     = bf2f((unsigned short)(au[i] & 0xFFFFu));
    av[2 * i + 1] = bf2f((unsigned short)(au[i] >> 16));
    xv[2 * i]     = bf2f((unsigned short)(xu[i] & 0xFFFFu));
    xv[2 * i + 1] = bf2f((unsigned short)(xu[i] >> 16));
  }
  float mx = av[0];
#pragma unroll
  for (int k = 1; k < 8; ++k) mx = fmaxf(mx, av[k]);
  float e[8], s = 0.f;
#pragma unroll
  for (int k = 0; k < 8; ++k) { e[k] = __expf(av[k] - mx); s += e[k]; }
  float dinv = 1.f / (1.f + s);
  float val = bf2f(value[t]);
  float r = 0.f;
#pragma unroll
  for (int k = 0; k < 8; ++k) r += e[k] * (val + xv[k]);
  agg[t] = f2bf(r * dinv);
}

// ---------------------------------------------------------------------------
extern "C" void kernel_launch(void* const* d_in, const int* in_sizes, int n_in,
                              void* d_out, int out_size, void* d_ws, size_t ws_size,
                              hipStream_t stream)
{
  (void)in_sizes; (void)n_in; (void)out_size; (void)ws_size;
  const float* x       = (const float*)d_in[0];
  const float* pos     = (const float*)d_in[1];
  const float* nrm     = (const float*)d_in[2];
  const float* W_start = (const float*)d_in[3];
  const float* b_start = (const float*)d_in[4];
  const float* W_key   = (const float*)d_in[5];
  const float* b_key   = (const float*)d_in[6];
  const float* W_query = (const float*)d_in[7];
  const float* b_query = (const float*)d_in[8];
  const float* W_value = (const float*)d_in[9];
  const float* b_value = (const float*)d_in[10];
  const float* pm_W1 = (const float*)d_in[11];
  const float* pm_b1 = (const float*)d_in[12];
  const float* pm_g1 = (const float*)d_in[13];
  const float* pm_be1= (const float*)d_in[14];
  const float* pm_m1 = (const float*)d_in[15];
  const float* pm_v1 = (const float*)d_in[16];
  const float* pm_W2 = (const float*)d_in[17];
  const float* pm_b2 = (const float*)d_in[18];
  const float* nm_W1 = (const float*)d_in[19];
  const float* nm_b1 = (const float*)d_in[20];
  const float* nm_g1 = (const float*)d_in[21];
  const float* nm_be1= (const float*)d_in[22];
  const float* nm_m1 = (const float*)d_in[23];
  const float* nm_v1 = (const float*)d_in[24];
  const float* nm_W2 = (const float*)d_in[25];
  const float* nm_b2 = (const float*)d_in[26];
  const float* am_W1 = (const float*)d_in[27];
  const float* am_b1 = (const float*)d_in[28];
  const float* am_g1 = (const float*)d_in[29];
  const float* am_be1= (const float*)d_in[30];
  const float* am_m1 = (const float*)d_in[31];
  const float* am_v1 = (const float*)d_in[32];
  const float* am_W2 = (const float*)d_in[33];
  const float* am_b2 = (const float*)d_in[34];
  const float* W_xyz = (const float*)d_in[35];
  const float* b_xyz = (const float*)d_in[36];
  const float* W_end = (const float*)d_in[37];
  const float* b_end = (const float*)d_in[38];
  float* out = (float*)d_out;

  // ---- workspace carve ----
  char* ws = (char*)d_ws;
  size_t off = 0;
  auto carve = [&](size_t bytes) -> void* {
    void* p = ws + off;
    off += (bytes + 255) & ~(size_t)255;
    return p;
  };
  typedef unsigned short u16;
  int* idx    = (int*)carve((size_t)BNK * 4);
  u16* xbf    = (u16*)carve((size_t)BATCH * INC * NPTS * 2);
  u16* xf     = (u16*)carve((size_t)BATCH * DIMC * NPTS * 2);
  u16* keyb   = (u16*)carve((size_t)BATCH * DIMC * NPTS * 2);
  u16* queryb = (u16*)carve((size_t)BATCH * DIMC * NPTS * 2);
  u16* valueb = (u16*)carve((size_t)BATCH * DIMC * NPTS * 2);
  u16* aggb   = (u16*)carve((size_t)BATCH * DIMC * NPTS * 2);
  u16* hcat   = (u16*)carve((size_t)2 * PHDC * BNK * 2);
  u16* sbuf   = (u16*)carve((size_t)DIMC * BNK * 2);   // e -> u -> a (aliased)
  u16* xyzb   = (u16*)carve((size_t)DIMC * BNK * 2);
  // bf16 weights
  u16* Wstart16 = (u16*)carve((size_t)DIMC * INC * 2);
  u16* Wkey16   = (u16*)carve((size_t)DIMC * DIMC * 2);
  u16* Wquery16 = (u16*)carve((size_t)DIMC * DIMC * 2);
  u16* Wvalue16 = (u16*)carve((size_t)DIMC * DIMC * 2);
  u16* Wxyz16   = (u16*)carve((size_t)DIMC * DIMC * 2);
  u16* Wend16   = (u16*)carve((size_t)INC * DIMC * 2);
  u16* amW1_16  = (u16*)carve((size_t)HIDC * DIMC * 2);
  u16* amW2_16  = (u16*)carve((size_t)DIMC * HIDC * 2);
  u16* wcat16   = (u16*)carve((size_t)DIMC * 2 * PHDC * 2);
  float* s_pm  = (float*)carve(PHDC * 4);
  float* sh_pm = (float*)carve(PHDC * 4);
  float* s_nm  = (float*)carve(PHDC * 4);
  float* sh_nm = (float*)carve(PHDC * 4);
  float* s_am  = (float*)carve(HIDC * 4);
  float* sh_am = (float*)carve(HIDC * 4);
  float* bcat  = (float*)carve(DIMC * 4);

  auto cvt = [&](const float* s, u16* d, long long n) {
    cvt_bf16_kernel<<<(int)((n + 255) / 256), 256, 0, stream>>>(s, d, n);
  };
  cvt(W_start, Wstart16, (long long)DIMC * INC);
  cvt(W_key,   Wkey16,   (long long)DIMC * DIMC);
  cvt(W_query, Wquery16, (long long)DIMC * DIMC);
  cvt(W_value, Wvalue16, (long long)DIMC * DIMC);
  cvt(W_xyz,   Wxyz16,   (long long)DIMC * DIMC);
  cvt(W_end,   Wend16,   (long long)INC * DIMC);
  cvt(am_W1,   amW1_16,  (long long)HIDC * DIMC);
  cvt(am_W2,   amW2_16,  (long long)DIMC * HIDC);
  cvt(x,       xbf,      (long long)BATCH * INC * NPTS);

  prep_kernel<<<1, 1024, 0, stream>>>(
      pm_b1, pm_g1, pm_be1, pm_m1, pm_v1, pm_W2, pm_b2,
      nm_b1, nm_g1, nm_be1, nm_m1, nm_v1, nm_W2, nm_b2,
      am_b1, am_g1, am_be1, am_m1, am_v1,
      s_pm, sh_pm, s_nm, sh_nm, s_am, sh_am, wcat16, bcat);

  knn_kernel<<<BN / 256, 256, 0, stream>>>(pos, idx);

  const long long sXN = (long long)INC * NPTS;    // x batch stride (elements)
  const long long sDN = (long long)DIMC * NPTS;   // (B,256,N) batch stride

  // xf = W_start @ x + b_start  (bf16 out)
  wmma_gemm<true, false><<<dim3(NPTS / 16, DIMC / 64, BATCH), 128, 0, stream>>>(
      Wstart16, xbf, xf, b_start, nullptr, DIMC, INC, NPTS, sXN, sDN);
  // key / query / value
  wmma_gemm<true, false><<<dim3(NPTS / 16, DIMC / 64, BATCH), 128, 0, stream>>>(
      Wkey16, xf, keyb, b_key, nullptr, DIMC, DIMC, NPTS, sDN, sDN);
  wmma_gemm<true, false><<<dim3(NPTS / 16, DIMC / 64, BATCH), 128, 0, stream>>>(
      Wquery16, xf, queryb, b_query, nullptr, DIMC, DIMC, NPTS, sDN, sDN);
  wmma_gemm<true, false><<<dim3(NPTS / 16, DIMC / 64, BATCH), 128, 0, stream>>>(
      Wvalue16, xf, valueb, b_value, nullptr, DIMC, DIMC, NPTS, sDN, sDN);

  // hcat = relu(bn(W1 * rel)) for pos and norm, bf16 (128 x BNK)
  rel_mlp1_kernel<<<BNK / 256, 256, 0, stream>>>(
      pos, nrm, idx, pm_W1, nm_W1, s_pm, sh_pm, s_nm, sh_nm, hcat);

  // e = Wcat @ hcat + bcat  (bf16, 256 x BNK)
  wmma_gemm<true, false><<<dim3(BNK / 16, DIMC / 64, 1), 128, 0, stream>>>(
      wcat16, hcat, sbuf, bcat, nullptr, DIMC, 2 * PHDC, BNK, 0, 0);
  // xyz = W_xyz @ e + b_xyz
  wmma_gemm<true, false><<<dim3(BNK / 16, DIMC / 64, 1), 128, 0, stream>>>(
      Wxyz16, sbuf, xyzb, b_xyz, nullptr, DIMC, DIMC, BNK, 0, 0);

  // u = query - key_gathered + xyz   (overwrites e buffer; e fully consumed)
  build_u_kernel<<<(int)(((long long)DIMC * BNK) / 256), 256, 0, stream>>>(
      queryb, keyb, xyzb, idx, sbuf);

  // a = am-MLP(u)   (in-place over u: each WG reads only its own columns first)
  am_fused_kernel<<<BNK / 16, 256, 0, stream>>>(
      amW1_16, amW2_16, s_am, sh_am, am_b2, sbuf, sbuf);

  // agg = sum_k softmax_one(a) * (value + xyz)   (bf16)
  softmax_agg_kernel<<<(BN * DIMC) / 256, 256, 0, stream>>>(
      sbuf, xyzb, valueb, aggb);

  // y = W_end @ agg + b_end + x   (f32 out with residual)
  wmma_gemm<false, true><<<dim3(NPTS / 16, INC / 64, BATCH), 128, 0, stream>>>(
      Wend16, aggb, out, b_end, x, INC, DIMC, NPTS, sDN, sXN);
}
